// CrossModalAttention_14164802142781
// MI455X (gfx1250) — matrix operand order repaired and checked by reading, not compiled, not run on hardware
//
#include <hip/hip_runtime.h>

#define DIMC   1024
#define HEADS  16
#define HD     64
#define SEQ    2048
#define BATCH  4
#define MROWS  (BATCH*SEQ)   // 8192
#define ATT_SCALE 0.125f     // 1/sqrt(64)

typedef __attribute__((ext_vector_type(16))) __bf16         v16bf;
typedef __attribute__((ext_vector_type(8)))  float          v8f;
typedef __attribute__((ext_vector_type(4)))  unsigned int   v4u;
typedef __attribute__((ext_vector_type(8)))  unsigned int   v8u;
typedef __attribute__((ext_vector_type(4)))  unsigned short v4us;

__device__ __forceinline__ unsigned short f32_to_bf16(float f) {
  unsigned int u = __builtin_bit_cast(unsigned int, f);
  u += 0x7FFFu + ((u >> 16) & 1u);            // round-to-nearest-even
  return (unsigned short)(u >> 16);
}

__device__ __forceinline__ v16bf make_frag(v4u lo, v4u hi) {
  v8u t;
  t[0]=lo[0]; t[1]=lo[1]; t[2]=lo[2]; t[3]=lo[3];
  t[4]=hi[0]; t[5]=hi[1]; t[6]=hi[2]; t[7]=hi[3];
  return __builtin_bit_cast(v16bf, t);
}

__device__ __forceinline__ v8f wmma_bf16(v16bf a, v16bf b, v8f c) {
  // D = A(16x32 bf16) * B(32x16 bf16) + C(16x16 f32)
  return __builtin_amdgcn_wmma_f32_16x16x32_bf16(false, a, false, b,
                                                 (short)0, c, false, false);
}

// ---- gfx1250 async global->LDS copy (ASYNCcnt path, bypasses VGPRs) ------
__device__ __forceinline__ unsigned lds_off32(const void* p) {
  // generic LDS pointer: addr[31:0] is the wave-relative LDS byte offset
  return (unsigned)(unsigned long long)p;
}
__device__ __forceinline__ void async_ld_b128(unsigned lds_byte_off, const void* gptr) {
  asm volatile("global_load_async_to_lds_b128 %0, %1, off"
               :: "v"(lds_byte_off), "v"((unsigned long long)gptr) : "memory");
}
__device__ __forceinline__ void wait_async_le4() {
  asm volatile("s_wait_asynccnt 0x4" ::: "memory");
}
__device__ __forceinline__ void wait_async_le2() {
  asm volatile("s_wait_asynccnt 0x2" ::: "memory");
}
__device__ __forceinline__ void wait_async_le0() {
  asm volatile("s_wait_asynccnt 0x0" ::: "memory");
}

// ---------------------------------------------------------------- convert
__global__ __launch_bounds__(256)
void cvt_f32_bf16(const float* __restrict__ in, unsigned short* __restrict__ out, int n) {
  int i = (blockIdx.x * 256 + threadIdx.x) * 4;
  if (i + 3 < n) {
    float4 v = *(const float4*)(in + i);
    v4us o;
    o[0] = f32_to_bf16(v.x); o[1] = f32_to_bf16(v.y);
    o[2] = f32_to_bf16(v.z); o[3] = f32_to_bf16(v.w);
    *(v4us*)(out + i) = o;
  }
}

// ---------------------------------------------------------------- GEMM
// C[M,N] = A[M,K] * W[N,K]^T + bias ; A,W bf16 ; out bf16 or f32.
// 128x128 block tile, 8 waves; async double-buffered 32-deep K staging.
__global__ __launch_bounds__(256)
void gemm_bf16_wmma(const unsigned short* __restrict__ A,
                    const unsigned short* __restrict__ W,
                    const float* __restrict__ bias,
                    unsigned short* __restrict__ out_bf16,
                    float* __restrict__ out_f32,
                    int M, int N, int K) {
  __shared__ __attribute__((aligned(16))) unsigned short lds_a[2][128 * 32];
  __shared__ __attribute__((aligned(16))) unsigned short lds_w[2][128 * 32];

  const int tid  = threadIdx.x;
  const int lane = tid & 31;
  const int wave = tid >> 5;
  const int nblk = N / 128;
  const int m0 = (blockIdx.x / nblk) * 128;
  const int n0 = (blockIdx.x % nblk) * 128;
  const int wm = (wave >> 2) * 64;   // wave grid 2 (M) x 4 (N)
  const int wn = (wave & 3) * 32;

  const v8f vzero = {0.f,0.f,0.f,0.f,0.f,0.f,0.f,0.f};
  v8f acc[4][2];
#pragma unroll
  for (int mt = 0; mt < 4; ++mt)
#pragma unroll
    for (int nt = 0; nt < 2; ++nt) acc[mt][nt] = vzero;

  const int srow = tid >> 2;        // 0..63 staging row
  const int sseg = (tid & 3) * 8;   // ushort offset in 32-wide k chunk
  const int lq    = lane & 15;
  const int khalf = (lane >> 4) * 8;   // A-frag K half
  const int kb16  = (lane >> 4) * 16;  // unused here; W frags use khalf

  auto stage = [&](int k0, int buf) {
    async_ld_b128(lds_off32(&lds_a[buf][srow * 32 + sseg]),
                  &A[(size_t)(m0 + srow) * K + k0 + sseg]);
    async_ld_b128(lds_off32(&lds_a[buf][(srow + 64) * 32 + sseg]),
                  &A[(size_t)(m0 + srow + 64) * K + k0 + sseg]);
    async_ld_b128(lds_off32(&lds_w[buf][srow * 32 + sseg]),
                  &W[(size_t)(n0 + srow) * K + k0 + sseg]);
    async_ld_b128(lds_off32(&lds_w[buf][(srow + 64) * 32 + sseg]),
                  &W[(size_t)(n0 + srow + 64) * K + k0 + sseg]);
  };

  const int iters = K / 32;
  stage(0, 0);                              // prefetch tile 0
  for (int i = 0; i < iters; ++i) {
    if (i + 1 < iters) { stage((i + 1) * 32, (i + 1) & 1); wait_async_le4(); }
    else               { wait_async_le0(); }
    __syncthreads();                        // tile i visible to all waves

    const unsigned short* la = lds_a[i & 1];
    const unsigned short* lw = lds_w[i & 1];
    v16bf af[4], wf[2];
#pragma unroll
    for (int mt = 0; mt < 4; ++mt) {
      const unsigned short* p = &la[(wm + mt * 16 + lq) * 32 + khalf];
      af[mt] = make_frag(*(const v4u*)p, *(const v4u*)(p + 16));
    }
#pragma unroll
    for (int nt = 0; nt < 2; ++nt) {
      const unsigned short* p = &lw[(wn + nt * 16 + lq) * 32 + khalf];
      wf[nt] = make_frag(*(const v4u*)p, *(const v4u*)(p + 16));
    }
#pragma unroll
    for (int mt = 0; mt < 4; ++mt)
#pragma unroll
      for (int nt = 0; nt < 2; ++nt)
        acc[mt][nt] = wmma_bf16(af[mt], wf[nt], acc[mt][nt]);

    __syncthreads();                        // reads done before buf reuse
  }

  const int hi8 = (lane >> 4) * 8;
#pragma unroll
  for (int nt = 0; nt < 2; ++nt) {
    const int col = n0 + wn + nt * 16 + lq;
    const float bv = bias ? bias[col] : 0.0f;
#pragma unroll
    for (int mt = 0; mt < 4; ++mt) {
#pragma unroll
      for (int r = 0; r < 8; ++r) {
        const int row = m0 + wm + mt * 16 + hi8 + r;
        const float v = acc[mt][nt][r] + bv;
        if (out_f32) out_f32[(size_t)row * N + col] = v;
        else         out_bf16[(size_t)row * N + col] = f32_to_bf16(v);
      }
    }
  }
}

// ---------------------------------------------------------------- attention
// One block per (b, h, 128-query chunk); 8 waves, each owns a 16-row q tile.
// K/V tiles async double-buffered in LDS and shared by all 8 waves.
__global__ __launch_bounds__(256)
void attn_wmma(const unsigned short* __restrict__ Qm,
               const unsigned short* __restrict__ Km,
               const unsigned short* __restrict__ Vm,
               float* __restrict__ Om) {
  __shared__ __attribute__((aligned(16))) unsigned short lds_k[2][32 * 64];
  __shared__ __attribute__((aligned(16))) unsigned short lds_v[2][32 * 64];
  __shared__ __attribute__((aligned(16))) unsigned short lds_p[8 * 16 * 32];

  const int tid  = threadIdx.x;
  const int lane = tid & 31;
  const int wave = tid >> 5;

  const int qchunks = SEQ / 128;              // 16
  const int qc = blockIdx.x % qchunks;
  const int h  = (blockIdx.x / qchunks) % HEADS;
  const int b  = blockIdx.x / (qchunks * HEADS);
  const int q0 = qc * 128 + wave * 16;
  const size_t rowbase = (size_t)b * SEQ;
  const size_t hoff = (size_t)h * HD;

  const int lq    = lane & 15;
  const int khalf = (lane >> 4) * 8;
  const int kb16  = (lane >> 4) * 16;
  const int hi8   = (lane >> 4) * 8;

  // Q A-fragments, d split into two 32-chunks
  v16bf qf[2];
  {
    const unsigned short* qp = Qm + (rowbase + q0 + lq) * DIMC + hoff;
#pragma unroll
    for (int c = 0; c < 2; ++c)
      qf[c] = make_frag(*(const v4u*)(qp + c * 32 + khalf),
                        *(const v4u*)(qp + c * 32 + 16 + khalf));
  }

  float m_r[8], l_r[8];
  v8f o_acc[4];
  const v8f vzero = {0.f,0.f,0.f,0.f,0.f,0.f,0.f,0.f};
#pragma unroll
  for (int r = 0; r < 8; ++r) { m_r[r] = -1e30f; l_r[r] = 0.0f; }
#pragma unroll
  for (int t = 0; t < 4; ++t) o_acc[t] = vzero;

  const int srow = tid >> 3;          // 0..31 staging row (one b128 per thread)
  const int sseg = (tid & 7) * 8;
  unsigned short* pbase = &lds_p[wave * 16 * 32];

  auto stageKV = [&](int kb, int buf) {
    async_ld_b128(lds_off32(&lds_k[buf][srow * 64 + sseg]),
                  &Km[(rowbase + kb + srow) * DIMC + hoff + sseg]);
    async_ld_b128(lds_off32(&lds_v[buf][srow * 64 + sseg]),
                  &Vm[(rowbase + kb + srow) * DIMC + hoff + sseg]);
  };

  const int iters = SEQ / 32;         // 64
  stageKV(0, 0);
  for (int i = 0; i < iters; ++i) {
    if (i + 1 < iters) { stageKV((i + 1) * 32, (i + 1) & 1); wait_async_le2(); }
    else               { wait_async_le0(); }
    __syncthreads();                  // K/V tile i visible

    const unsigned short* lk = lds_k[i & 1];
    const unsigned short* lv = lds_v[i & 1];

    // scores: two 16x16 tiles over this 32-key chunk
    v8f s[2];
#pragma unroll
    for (int j = 0; j < 2; ++j) {
      s[j] = vzero;
#pragma unroll
      for (int c = 0; c < 2; ++c) {
        const unsigned short* kp = &lk[(j * 16 + lq) * 64 + c * 32 + khalf];
        v16bf kf = make_frag(*(const v4u*)kp, *(const v4u*)(kp + 16));
        s[j] = wmma_bf16(qf[c], kf, s[j]);
      }
    }

    // online softmax; rows live in 16-lane halves of the wave
#pragma unroll
    for (int r = 0; r < 8; ++r) {
      float s0 = s[0][r] * ATT_SCALE;
      float s1 = s[1][r] * ATT_SCALE;
      float rm = fmaxf(s0, s1);
#pragma unroll
      for (int msk = 8; msk; msk >>= 1) rm = fmaxf(rm, __shfl_xor(rm, msk, 16));
      const float mnew  = fmaxf(m_r[r], rm);
      const float alpha = __expf(m_r[r] - mnew);
      const float p0 = __expf(s0 - mnew);
      const float p1 = __expf(s1 - mnew);
      float rs = p0 + p1;
#pragma unroll
      for (int msk = 8; msk; msk >>= 1) rs += __shfl_xor(rs, msk, 16);
      l_r[r] = l_r[r] * alpha + rs;
      m_r[r] = mnew;
#pragma unroll
      for (int t = 0; t < 4; ++t) o_acc[t][r] = o_acc[t][r] * alpha;
      const int prow = r + hi8;
      pbase[prow * 32 + lq]      = f32_to_bf16(p0);
      pbase[prow * 32 + 16 + lq] = f32_to_bf16(p1);
    }

    // P as A-fragment (16x32 over this key chunk)
    const unsigned short* pp = pbase + lq * 32 + kb16;
    v16bf pf = make_frag(*(const v4u*)pp, *(const v4u*)(pp + 8));

    // V B-fragments (32 keys x 16 d) gathered from LDS, 4 d-tiles
#pragma unroll
    for (int t = 0; t < 4; ++t) {
      const int ncol = t * 16 + lq;
      v8u vv;
#pragma unroll
      for (int e = 0; e < 8; ++e) {
        unsigned int vlo = lv[(kb16 + 2 * e)     * 64 + ncol];
        unsigned int vhi = lv[(kb16 + 2 * e + 1) * 64 + ncol];
        vv[e] = vlo | (vhi << 16);
      }
      o_acc[t] = wmma_bf16(pf, __builtin_bit_cast(v16bf, vv), o_acc[t]);
    }

    __syncthreads();                  // reads done before K/V buf reuse
  }

  // normalize + store fp32 into [B*N, DIM] at head column offset
#pragma unroll
  for (int r = 0; r < 8; ++r) {
    const float inv = 1.0f / l_r[r];
    const size_t row = rowbase + q0 + r + hi8;
#pragma unroll
    for (int t = 0; t < 4; ++t)
      Om[row * DIMC + hoff + t * 16 + lq] = o_acc[t][r] * inv;
  }
}

// ---------------------------------------------------------------- add + LN
__global__ __launch_bounds__(256)
void ln_fuse(const float* __restrict__ xa, const float* __restrict__ xb,
             const float* __restrict__ gamma, const float* __restrict__ beta,
             unsigned short* __restrict__ out) {
  __shared__ float ws1[8], ws2[8];
  const int row = blockIdx.x;
  const int tid = threadIdx.x;
  const size_t off = (size_t)row * DIMC + tid * 4;
  float4 a = *(const float4*)(xa + off);
  float4 c = *(const float4*)(xb + off);
  float x[4] = {a.x + c.x, a.y + c.y, a.z + c.z, a.w + c.w};
  float s = x[0] + x[1] + x[2] + x[3];
  float s2 = x[0]*x[0] + x[1]*x[1] + x[2]*x[2] + x[3]*x[3];
#pragma unroll
  for (int msk = 16; msk; msk >>= 1) {
    s  += __shfl_xor(s,  msk, 32);
    s2 += __shfl_xor(s2, msk, 32);
  }
  if ((tid & 31) == 0) { ws1[tid >> 5] = s; ws2[tid >> 5] = s2; }
  __syncthreads();
  float S = 0.f, S2 = 0.f;
#pragma unroll
  for (int w = 0; w < 8; ++w) { S += ws1[w]; S2 += ws2[w]; }
  const float mu  = S * (1.0f / DIMC);
  const float var = S2 * (1.0f / DIMC) - mu * mu;
  const float inv = rsqrtf(var + 1e-5f);
  v4us o;
#pragma unroll
  for (int j = 0; j < 4; ++j) {
    const int col = tid * 4 + j;
    o[j] = f32_to_bf16((x[j] - mu) * inv * gamma[col] + beta[col]);
  }
  *(v4us*)(out + off) = o;
}

// ---------------------------------------------------------------- launch
extern "C" void kernel_launch(void* const* d_in, const int* in_sizes, int n_in,
                              void* d_out, int out_size, void* d_ws, size_t ws_size,
                              hipStream_t stream) {
  const float* f_a = (const float*)d_in[0];
  const float* f_b = (const float*)d_in[1];
  // names: q_a, k_a, v_a, q_b, k_b, v_b, proj -> W at 2+2i, b at 3+2i
  const float* Wf[7];  const float* bf[7];
  for (int i = 0; i < 7; ++i) { Wf[i] = (const float*)d_in[2 + 2 * i]; bf[i] = (const float*)d_in[3 + 2 * i]; }
  const float* ln_g = (const float*)d_in[16];
  const float* ln_b = (const float*)d_in[17];

  const size_t S2 = (size_t)MROWS * DIMC * sizeof(unsigned short); // 16 MiB act
  const size_t W2 = (size_t)DIMC * DIMC * sizeof(unsigned short);  //  2 MiB weight
  char* w = (char*)d_ws;
  unsigned short* fa16 = (unsigned short*)(w);
  unsigned short* fb16 = (unsigned short*)(w + S2);
  unsigned short* wt16[7];
  for (int i = 0; i < 7; ++i) wt16[i] = (unsigned short*)(w + 2 * S2 + (size_t)i * W2);
  char* actb = w + 2 * S2 + 7 * W2;
  unsigned short* qa16 = (unsigned short*)(actb + 0 * S2);
  unsigned short* ka16 = (unsigned short*)(actb + 1 * S2);
  unsigned short* va16 = (unsigned short*)(actb + 2 * S2);
  unsigned short* qb16 = (unsigned short*)(actb + 3 * S2);
  unsigned short* kb16 = (unsigned short*)(actb + 4 * S2);
  unsigned short* vb16 = (unsigned short*)(actb + 5 * S2);
  // overlap-safe reuse: fa/fb dead after projections; kb/vb dead after attn1; qb dead after attn2
  float* fta = (float*)(w);               // 32 MiB over fa16+fb16
  float* ftb = (float*)(actb + 4 * S2);   // 32 MiB over kb16+vb16
  unsigned short* fused16 = (unsigned short*)(actb + 3 * S2); // over qb16

  const int nAct = MROWS * DIMC;   // 8,388,608
  const int nW   = DIMC * DIMC;    // 1,048,576
  // 1) convert to bf16
  cvt_f32_bf16<<<nAct / 1024, 256, 0, stream>>>(f_a, fa16, nAct);
  cvt_f32_bf16<<<nAct / 1024, 256, 0, stream>>>(f_b, fb16, nAct);
  for (int i = 0; i < 7; ++i)
    cvt_f32_bf16<<<nW / 1024, 256, 0, stream>>>(Wf[i], wt16[i], nW);

  // 2) six QKV projections (bf16 out)
  const dim3 ggrid((MROWS / 128) * (DIMC / 128));   // 512
  gemm_bf16_wmma<<<ggrid, 256, 0, stream>>>(fa16, wt16[0], bf[0], qa16, nullptr, MROWS, DIMC, DIMC);
  gemm_bf16_wmma<<<ggrid, 256, 0, stream>>>(fa16, wt16[1], bf[1], ka16, nullptr, MROWS, DIMC, DIMC);
  gemm_bf16_wmma<<<ggrid, 256, 0, stream>>>(fa16, wt16[2], bf[2], va16, nullptr, MROWS, DIMC, DIMC);
  gemm_bf16_wmma<<<ggrid, 256, 0, stream>>>(fb16, wt16[3], bf[3], qb16, nullptr, MROWS, DIMC, DIMC);
  gemm_bf16_wmma<<<ggrid, 256, 0, stream>>>(fb16, wt16[4], bf[4], kb16, nullptr, MROWS, DIMC, DIMC);
  gemm_bf16_wmma<<<ggrid, 256, 0, stream>>>(fb16, wt16[5], bf[5], vb16, nullptr, MROWS, DIMC, DIMC);

  // 3) cross attentions
  const dim3 agrid(BATCH * HEADS * (SEQ / 128));    // 1024
  attn_wmma<<<agrid, 256, 0, stream>>>(qa16, kb16, vb16, fta);
  attn_wmma<<<agrid, 256, 0, stream>>>(qb16, ka16, va16, ftb);

  // 4) add + layernorm -> bf16
  ln_fuse<<<MROWS, 256, 0, stream>>>(fta, ftb, ln_g, ln_b, fused16);

  // 5) output projection (fp32 out)
  gemm_bf16_wmma<<<ggrid, 256, 0, stream>>>(fused16, wt16[6], bf[6], nullptr, (float*)d_out, MROWS, DIMC, DIMC);
}
